// HierarchicalRVQ_23398981829014
// MI455X (gfx1250) — matrix operand order repaired and checked by reading, not compiled, not run on hardware
//
#include <hip/hip_runtime.h>
#include <hip/hip_bf16.h>
#include <math.h>
#include <float.h>

#define BB 16
#define DD 512
#define TT 4096
#define NN (BB * TT)   // 65536 rows
#define HID 32
#define CWf 0.25f
#define GR 64          // residual rows per GEMM workgroup (2 waves x 2 M-tiles x 16)
#define LDS_PITCH 520  // bf16 elements per row in LDS (conflict-free A-frag reads)

typedef __attribute__((ext_vector_type(16))) __bf16 v16bf;
typedef __attribute__((ext_vector_type(8)))  __bf16 v8bf;
typedef __attribute__((ext_vector_type(4)))  __bf16 v4bf;
typedef __attribute__((ext_vector_type(8)))  float  v8f;
typedef __attribute__((ext_vector_type(4)))  int    v4i;

typedef __attribute__((address_space(1))) v4i gv4i;   // global <4 x i32>
typedef __attribute__((address_space(3))) v4i lv4i;   // LDS    <4 x i32>

union ABfrag { v16bf v; v8bf h[2]; };

// ---------------------------------------------------------------------------
// gfx1250 async global->LDS copy (16B), with portable fallback
// ---------------------------------------------------------------------------
__device__ __forceinline__ void async_copy_b128(const void* g, void* l) {
#if __has_builtin(__builtin_amdgcn_global_load_async_to_lds_b128)
  __builtin_amdgcn_global_load_async_to_lds_b128((gv4i*)g, (lv4i*)l, 0, 0);
#else
  *(v8bf*)l = *(const v8bf*)g;
#endif
}

__device__ __forceinline__ void wait_async_zero() {
  asm volatile("s_wait_asynccnt 0x0" ::: "memory");
}

// ---------------------------------------------------------------------------
// x [B,D,T] -> RF f32 [N,D] and RB bf16 [N,D] (N = B*T), tiled LDS transpose
// ---------------------------------------------------------------------------
__global__ void k_transpose_in(const float* __restrict__ x, float* __restrict__ RF,
                               __bf16* __restrict__ RB) {
  __shared__ float tile[32][33];
  int tx = threadIdx.x, ty = threadIdx.y;
  int t0 = blockIdx.x * 32, d0 = blockIdx.y * 32, b = blockIdx.z;
#pragma unroll
  for (int k = 0; k < 4; ++k) {
    int d = d0 + ty * 4 + k;
    tile[ty * 4 + k][tx] = x[((size_t)b * DD + d) * TT + t0 + tx];  // coalesced in t
  }
  __syncthreads();
#pragma unroll
  for (int k = 0; k < 4; ++k) {
    int t = t0 + ty * 4 + k;
    float v = tile[tx][ty * 4 + k];
    size_t o = ((size_t)b * TT + t) * DD + d0 + tx;                 // coalesced in d
    RF[o] = v;
    RB[o] = (__bf16)v;
  }
}

// ---------------------------------------------------------------------------
// embed f32 [K,D] -> bf16 EB [K,D]; hn[k] = 0.5*||e_k||^2; zero hist / closs
// ---------------------------------------------------------------------------
__global__ void k_embed_prep(const float* __restrict__ E, int K,
                             __bf16* __restrict__ EB, float* __restrict__ hn,
                             int* __restrict__ hist, float* __restrict__ closs_s) {
  int tid = threadIdx.x, lane = tid & 31, w = tid >> 5;
  int k = blockIdx.x * 8 + w;
  if (k < K) {
    float s = 0.f;
    for (int d = lane; d < DD; d += 32) {
      float v = E[(size_t)k * DD + d];
      EB[(size_t)k * DD + d] = (__bf16)v;
      s += v * v;
    }
#pragma unroll
    for (int m = 16; m >= 1; m >>= 1) s += __shfl_xor(s, m, 32);
    if (lane == 0) hn[k] = 0.5f * s;
  }
  int gid = blockIdx.x * blockDim.x + tid;
  if (gid < K) hist[gid] = 0;
  if (gid == 0) *closs_s = 0.f;
}

// ---------------------------------------------------------------------------
// Fused distance-GEMM + argmin.  scores = R @ E^T - 0.5||e||^2 via
// v_wmma_f32_16x16x32_bf16.  Each wave owns 2 M-tiles (32 rows) so every
// B fragment is reused twice (halves codebook L2 traffic).  The residual
// tile is staged bf16 -> LDS with async global->LDS b128 copies, scattered
// into a padded pitch for conflict-free A-fragment reads.
// ---------------------------------------------------------------------------
__global__ void __launch_bounds__(64)
k_gemm_argmax(const __bf16* __restrict__ RB, const __bf16* __restrict__ EB,
              const float* __restrict__ hn, int K,
              int* __restrict__ idx_ws, float* __restrict__ idx_out) {
  extern __shared__ __bf16 Rlds[];   // GR * LDS_PITCH bf16
  const int tid = threadIdx.x;
  const int r0 = blockIdx.x * GR;

  // stage GR x 512 bf16 tile: 16B chunks, per-lane LDS destinations
  for (int i = tid; i < GR * (DD / 8); i += 64) {   // 64 chunks per row
    int row = i >> 6;
    int c   = i & 63;
    async_copy_b128(RB + (size_t)(r0 + row) * DD + (c << 3),
                    &Rlds[row * LDS_PITCH + (c << 3)]);
  }
  wait_async_zero();
  __syncthreads();

  const int lane = tid & 31;
  const int wave = tid >> 5;
  const int lw   = lane & 15;
  const int half = lane >> 4;
  const __bf16* arow0 = &Rlds[(wave * 32 + lw) * LDS_PITCH];        // M-tile 0
  const __bf16* arow1 = arow0 + 16 * LDS_PITCH;                     // M-tile 1

  float bestv[2][8];
  int   besti[2][8];
#pragma unroll
  for (int mt = 0; mt < 2; ++mt)
#pragma unroll
    for (int j = 0; j < 8; ++j) { bestv[mt][j] = -FLT_MAX; besti[mt][j] = 0; }

  const int nct = K >> 4;
  for (int ct = 0; ct < nct; ++ct) {
    int col = (ct << 4) + lw;
    if (ct + 1 < nct) __builtin_prefetch(EB + (size_t)(col + 16) * DD, 0, 0);
    float hnc = hn[col];
    v8f acc0 = {0.f, 0.f, 0.f, 0.f, 0.f, 0.f, 0.f, 0.f};
    v8f acc1 = acc0;
    const __bf16* brow = EB + (size_t)col * DD + (half << 3);
#pragma unroll 4
    for (int kk = 0; kk < DD; kk += 32) {
      ABfrag bm;                                               // B frag (reused x2)
      bm.h[0] = *(const v8bf*)(brow + kk);
      bm.h[1] = *(const v8bf*)(brow + kk + 16);
      ABfrag a0, a1;
      a0.h[0] = *(const v8bf*)(arow0 + kk + (half << 3));
      a0.h[1] = *(const v8bf*)(arow0 + kk + 16 + (half << 3));
      a1.h[0] = *(const v8bf*)(arow1 + kk + (half << 3));
      a1.h[1] = *(const v8bf*)(arow1 + kk + 16 + (half << 3));
      acc0 = __builtin_amdgcn_wmma_f32_16x16x32_bf16(false, a0.v, false, bm.v,
                                                     (short)0, acc0, false, false);
      acc1 = __builtin_amdgcn_wmma_f32_16x16x32_bf16(false, a1.v, false, bm.v,
                                                     (short)0, acc1, false, false);
    }
#pragma unroll
    for (int j = 0; j < 8; ++j) {
      float s0 = acc0[j] - hnc;
      if (s0 > bestv[0][j]) { bestv[0][j] = s0; besti[0][j] = col; }
      float s1 = acc1[j] - hnc;
      if (s1 > bestv[1][j]) { bestv[1][j] = s1; besti[1][j] = col; }
    }
  }
  // reduce over the 16 lanes of each half (lanes 0-15: rows j, 16-31: rows j+8)
#pragma unroll
  for (int mt = 0; mt < 2; ++mt) {
#pragma unroll
    for (int j = 0; j < 8; ++j) {
#pragma unroll
      for (int m = 8; m >= 1; m >>= 1) {
        float ov = __shfl_xor(bestv[mt][j], m, 32);
        int   oi = __shfl_xor(besti[mt][j], m, 32);
        if (ov > bestv[mt][j] || (ov == bestv[mt][j] && oi < besti[mt][j])) {
          bestv[mt][j] = ov; besti[mt][j] = oi;   // argmin tie-break: smaller idx
        }
      }
    }
    if (lw == 0) {
      int rw = r0 + wave * 32 + mt * 16 + (half << 3);
#pragma unroll
      for (int j = 0; j < 8; ++j) {
        idx_ws[rw + j]  = besti[mt][j];
        idx_out[rw + j] = (float)besti[mt][j];
      }
    }
  }
}

// ---------------------------------------------------------------------------
// Fused gate MLP + gather + residual update + closs + histogram.
// One wave per row; lane j == hidden unit j (HID == wave32 width).
// ---------------------------------------------------------------------------
__global__ void __launch_bounds__(256)
k_apply(float* __restrict__ RF, __bf16* __restrict__ RB,
        const float* __restrict__ E, const int* __restrict__ idx_ws,
        const float* __restrict__ W1, const float* __restrict__ b1,
        const float* __restrict__ W2, const float* __restrict__ b2,
        int* __restrict__ hist, float* __restrict__ closs_s) {
  __shared__ float W1s[DD * HID];   // [d][j], exactly 64 KB, conflict-free reads
  const int tid = threadIdx.x;
  for (int i = tid; i < DD * HID; i += 256) {
    int d = i >> 5, j = i & 31;
    W1s[i] = W1[j * DD + d];
  }
  __syncthreads();

  const int lane = tid & 31;
  const int j = lane;
  const float b1j = b1[j];
  const float w2j = W2[j];
  const float b2v = b2[0];
  const int gw = blockIdx.x * 8 + (tid >> 5);
  const int nwaves = gridDim.x * 8;
  float cacc = 0.f;

  for (int n = gw; n < NN; n += nwaves) {
    const float* r = RF + (size_t)n * DD;
    float h = b1j;
    const float4* r4 = (const float4*)r;
    for (int c = 0; c < DD / 4; ++c) {
      float4 rv = r4[c];
      int dbase = c << 2;
      h += W1s[(dbase    ) * HID + j] * rv.x;
      h += W1s[(dbase + 1) * HID + j] * rv.y;
      h += W1s[(dbase + 2) * HID + j] * rv.z;
      h += W1s[(dbase + 3) * HID + j] * rv.w;
    }
    // tanh-approx GELU (jax.nn.gelu default)
    float g1 = 0.5f * h * (1.f + tanhf(0.7978845608028654f * (h + 0.044715f * h * h * h)));
    float lg = w2j * g1;
#pragma unroll
    for (int m = 16; m >= 1; m >>= 1) lg += __shfl_xor(lg, m, 32);
    float logit = lg + b2v;
    float soft  = 1.f / (1.f + __expf(-logit));
    float hard  = soft > 0.5f ? 1.f : 0.f;
    float g     = (hard + soft) - soft;      // straight-through forward value

    int idx = idx_ws[n];
    if (lane == 0) atomicAdd(&hist[idx], 1);
    const float* q = E + (size_t)idx * DD;
    for (int d = lane; d < DD; d += 32) {
      float rv = r[d];
      float qv = q[d];
      float df = rv - qv;                     // closs uses ungated q
      cacc += df * df;
      float nv = rv - g * qv;                 // residual -= gated q
      RF[(size_t)n * DD + d] = nv;
      RB[(size_t)n * DD + d] = (__bf16)nv;    // keep bf16 mirror for next GEMM
    }
  }
#pragma unroll
  for (int m = 16; m >= 1; m >>= 1) cacc += __shfl_xor(cacc, m, 32);
  if (lane == 0) atomicAdd(closs_s, cacc);
}

// ---------------------------------------------------------------------------
// Perplexities + commitment loss (single workgroup)
// ---------------------------------------------------------------------------
__global__ void k_stats(const int* __restrict__ hist, const float* __restrict__ closs_ws,
                        float* __restrict__ out_closs, float* __restrict__ out_perp) {
  __shared__ float red[1024];
  const int tid = threadIdx.x;
  const int Ks[3] = {1024, 512, 256};
  for (int s = 0; s < 3; ++s) {
    float e = 0.f;
    const int* h = hist + s * 1024;
    for (int k = tid; k < Ks[s]; k += 1024) {
      float p = (float)h[k] / (float)NN;
      e += p * logf(p + 1e-10f);
    }
    red[tid] = e;
    __syncthreads();
    for (int st = 512; st > 0; st >>= 1) {
      if (tid < st) red[tid] += red[tid + st];
      __syncthreads();
    }
    if (tid == 0) out_perp[s] = __expf(-red[0]);
    __syncthreads();
  }
  if (tid == 0)
    *out_closs = CWf * (closs_ws[0] + closs_ws[1] + closs_ws[2]) /
                 (float)((size_t)NN * DD);
}

// ---------------------------------------------------------------------------
// total[b,d,t] = x[b,d,t] - residual_final  (tiled transpose back)
// ---------------------------------------------------------------------------
__global__ void k_total(const float* __restrict__ x, const float* __restrict__ RF,
                        float* __restrict__ total) {
  __shared__ float tile[32][33];
  int tx = threadIdx.x, ty = threadIdx.y;
  int t0 = blockIdx.x * 32, d0 = blockIdx.y * 32, b = blockIdx.z;
#pragma unroll
  for (int k = 0; k < 4; ++k) {
    int t = t0 + ty * 4 + k;
    tile[tx][ty * 4 + k] = RF[((size_t)b * TT + t) * DD + d0 + tx]; // coalesced in d
  }
  __syncthreads();
#pragma unroll
  for (int k = 0; k < 4; ++k) {
    int d = d0 + ty * 4 + k;
    size_t o = ((size_t)b * DD + d) * TT + t0 + tx;
    total[o] = x[o] - tile[ty * 4 + k][tx];                          // coalesced in t
  }
}

// ---------------------------------------------------------------------------
extern "C" void kernel_launch(void* const* d_in, const int* in_sizes, int n_in,
                              void* d_out, int out_size, void* d_ws, size_t ws_size,
                              hipStream_t stream) {
  const float* x   = (const float*)d_in[0];
  const float* Es[3] = {(const float*)d_in[1], (const float*)d_in[2], (const float*)d_in[3]};
  const float* gw1 = (const float*)d_in[4];
  const float* gb1 = (const float*)d_in[5];
  const float* gw2 = (const float*)d_in[6];
  const float* gb2 = (const float*)d_in[7];
  float* out = (float*)d_out;

  const size_t OFF_CLOSS = (size_t)BB * DD * TT;   // 33554432
  const size_t OFF_PERP  = OFF_CLOSS + 1;
  const size_t OFF_IDX   = OFF_PERP + 3;

  char* ws = (char*)d_ws;
  float*  RF   = (float*)(ws);                                   // 128 MB f32 residual
  __bf16* RB   = (__bf16*)(ws + 134217728);                      // 64 MB bf16 mirror
  __bf16* EB   = (__bf16*)(ws + 134217728 + 67108864);           // 1 MB
  float*  hn   = (float*) (ws + 134217728 + 67108864 + 1048576);
  int*    idxw = (int*)   (ws + 134217728 + 67108864 + 1048576 + 4096);
  int*    hist = (int*)   (ws + 134217728 + 67108864 + 1048576 + 4096 + 262144);
  float*  clw  = (float*) (ws + 134217728 + 67108864 + 1048576 + 4096 + 262144 + 12288);

  const int Ks[3] = {1024, 512, 256};
  const size_t gemm_lds = (size_t)GR * LDS_PITCH * sizeof(__bf16);  // 66,560 B

  hipLaunchKernelGGL(k_transpose_in, dim3(TT / 32, DD / 32, BB), dim3(32, 8), 0, stream,
                     x, RF, RB);
  for (int s = 0; s < 3; ++s) {
    int K = Ks[s];
    hipLaunchKernelGGL(k_embed_prep, dim3((K + 7) / 8), dim3(256), 0, stream,
                       Es[s], K, EB, hn, hist + s * 1024, clw + s);
    hipLaunchKernelGGL(k_gemm_argmax, dim3(NN / GR), dim3(64), gemm_lds, stream,
                       RB, EB, hn, K, idxw, out + OFF_IDX + (size_t)s * NN);
    hipLaunchKernelGGL(k_apply, dim3(2048), dim3(256), 0, stream,
                       RF, RB, Es[s], idxw,
                       gw1 + (size_t)s * HID * DD, gb1 + (size_t)s * HID,
                       gw2 + (size_t)s * HID,      gb2 + s,
                       hist + s * 1024, clw + s);
  }
  hipLaunchKernelGGL(k_stats, dim3(1), dim3(1024), 0, stream,
                     hist, clw, out + OFF_CLOSS, out + OFF_PERP);
  hipLaunchKernelGGL(k_total, dim3(TT / 32, DD / 32, BB), dim3(32, 8), 0, stream,
                     x, RF, out);
}
